// LightweightMambaBlock_56968446214310
// MI455X (gfx1250) — matrix operand order repaired and checked by reading, not compiled, not run on hardware
//
#include <hip/hip_runtime.h>
#include <hip/hip_bf16.h>
#include <math.h>

// ---------------------------------------------------------------------------
// LightweightMambaBlock on MI455X (gfx1250), wave32 + WMMA bf16 path.
//   GEMM1: xz = x @ W_in^T          (16384x1024 @ 1024x3072)
//   conv+silu (depthwise k=3)
//   GEMM2: dt = softplus(xc @ dt_w^T + b); fused exp/scale epilogue
//   scan : chunked parallel cumsum over L per (b,c)  (3 phases)
//   GEMM3: out = y @ W_out^T        (16384x1536 @ 1536x1024)
// GEMMs: LDS-staged double-buffered block tile 64x128 (8 waves x 32x32),
// v_wmma_f32_16x16x32_bf16 with f32 accumulation; N,K compile-time.
// ---------------------------------------------------------------------------

typedef __attribute__((ext_vector_type(16))) __bf16 bf16x16;
typedef __attribute__((ext_vector_type(8)))  __bf16 bf16x8;
typedef __attribute__((ext_vector_type(8)))  float  f32x8;

#define DIMX   1024
#define DINNER 1536
#define DPROJ  3072          // 2*DINNER
#define LSEQ   4096
#define NBAT   4
#define NTOK   (NBAT * LSEQ) // 16384
#define CHUNK  128
#define NCHUNK (LSEQ / CHUNK) // 32

static __device__ __forceinline__ __bf16 f2bf(float f) {
  unsigned u = __builtin_bit_cast(unsigned, f);
  u += 0x7FFFu + ((u >> 16) & 1u);           // round-to-nearest-even
  unsigned short h = (unsigned short)(u >> 16);
  return __builtin_bit_cast(__bf16, h);
}

static __device__ __forceinline__ float silu(float v) {
  return v / (1.0f + expf(-v));
}

// A frag: 16 contiguous K elements at p
static __device__ __forceinline__ bf16x16 load_frag_a(const __bf16* p) {
  bf16x8 lo = *(const bf16x8*)p;
  bf16x8 hi = *(const bf16x8*)(p + 8);
  return __builtin_shufflevector(lo, hi, 0,1,2,3,4,5,6,7,8,9,10,11,12,13,14,15);
}
// B frag: 16 contiguous K elements at p
static __device__ __forceinline__ bf16x16 load_frag_b(const __bf16* p) {
  bf16x8 lo = *(const bf16x8*)p;
  bf16x8 hi = *(const bf16x8*)(p + 8);
  return __builtin_shufflevector(lo, hi, 0,1,2,3,4,5,6,7,8,9,10,11,12,13,14,15);
}

// ---------------------------------------------------------------- convert ---
__global__ void cvt_bf16_kernel(const float* __restrict__ in,
                                __bf16* __restrict__ out, int n) {
  int i = blockIdx.x * blockDim.x + threadIdx.x;
  if (i < n) out[i] = f2bf(in[i]);
}

// -------------------------------------------------- LDS-staged WMMA GEMM ---
// C[M,N] = A[M,K](bf16 rm) * Bw[N,K]^T (bf16, torch weight layout), M=16384.
// Block: 256 thr = 8 waves (2M x 4N), block tile 64x128, wave tile 32x32.
// K staged through LDS in KS=64 slices, double buffered. N,K compile-time.
template <int MODE, int N, int K>
__global__ __launch_bounds__(256) void wmma_gemm_lds(
    const __bf16* __restrict__ A,
    const __bf16* __restrict__ Bw,
    float* __restrict__ C,
    const float* __restrict__ bias,
    const float* __restrict__ Avec,
    const float* __restrict__ xconv,
    float* __restrict__ s_out,
    float* __restrict__ aexp_out) {
  constexpr int BM = 64, BN = 128, KS = 64;
  constexpr int LDK = KS + 8;                  // +16B pad: spread ds banks
  __shared__ __bf16 SA[2][BM * LDK];
  __shared__ __bf16 SB[2][BN * LDK];

  const int tid  = threadIdx.x;
  const int lane = tid & 31;
  const int wave = tid >> 5;
  const int wm   = wave & 1;                   // 32-row band in block
  const int wn   = wave >> 1;                  // 32-col band in block
  const int half = lane >> 4;
  const int l15  = lane & 15;
  const int akoff = half * 16;                 // A frag base (k-local)
  const int bkoff = half * 16;                 // B frag base (k-local)

  constexpr int MBLK = NTOK / BM;              // 256 (exact, no guards)
  const int mb0 = (blockIdx.x % MBLK) * BM;
  const int nb0 = (blockIdx.x / MBLK) * BN;

  const __bf16* Ab = A  + (size_t)mb0 * K;
  const __bf16* Bb = Bw + (size_t)nb0 * K;

  f32x8 acc[2][2] = {};

  // ---- stage 0 into LDS buffer 0 (global -> ds) ----
#pragma unroll
  for (int j = 0; j < 2; ++j) {                // A: 512 16B chunks / 256 thr
    int q = tid + j * 256, row = q >> 3, c8 = (q & 7) * 8;
    *(bf16x8*)&SA[0][row * LDK + c8] = *(const bf16x8*)(Ab + (size_t)row * K + c8);
  }
#pragma unroll
  for (int j = 0; j < 4; ++j) {                // B: 1024 16B chunks / 256 thr
    int q = tid + j * 256, row = q >> 3, c8 = (q & 7) * 8;
    *(bf16x8*)&SB[0][row * LDK + c8] = *(const bf16x8*)(Bb + (size_t)row * K + c8);
  }
  __syncthreads();

  for (int ks = 0; ks < K; ks += KS) {
    const int cur = (ks / KS) & 1;
    const int nxt = cur ^ 1;
    const bool more = (ks + KS) < K;

    // ---- issue next stage's global loads (held in regs; waits deferred) ----
    bf16x8 rA[2], rB[4];
    if (more) {
      __builtin_prefetch(Ab + ks + 2 * KS, 0, 1);          // global_prefetch_b8
      __builtin_prefetch(Bb + ks + 2 * KS, 0, 1);
#pragma unroll
      for (int j = 0; j < 2; ++j) {
        int q = tid + j * 256, row = q >> 3, c8 = (q & 7) * 8;
        rA[j] = *(const bf16x8*)(Ab + (size_t)row * K + ks + KS + c8);
      }
#pragma unroll
      for (int j = 0; j < 4; ++j) {
        int q = tid + j * 256, row = q >> 3, c8 = (q & 7) * 8;
        rB[j] = *(const bf16x8*)(Bb + (size_t)row * K + ks + KS + c8);
      }
    }

    // ---- compute current stage from LDS (2 K-steps of 32) ----
#pragma unroll
    for (int kk = 0; kk < KS; kk += 32) {
      const __bf16* pa = &SA[cur][(wm * 32 + l15) * LDK + kk + akoff];
      const __bf16* pb = &SB[cur][(wn * 32 + l15) * LDK + kk + bkoff];
      bf16x16 a0 = load_frag_a(pa);
      bf16x16 a1 = load_frag_a(pa + 16 * LDK);
      bf16x16 b0 = load_frag_b(pb);
      bf16x16 b1 = load_frag_b(pb + 16 * LDK);
      acc[0][0] = __builtin_amdgcn_wmma_f32_16x16x32_bf16(false, a0, false, b0, (short)0, acc[0][0], false, false);
      acc[0][1] = __builtin_amdgcn_wmma_f32_16x16x32_bf16(false, a0, false, b1, (short)0, acc[0][1], false, false);
      acc[1][0] = __builtin_amdgcn_wmma_f32_16x16x32_bf16(false, a1, false, b0, (short)0, acc[1][0], false, false);
      acc[1][1] = __builtin_amdgcn_wmma_f32_16x16x32_bf16(false, a1, false, b1, (short)0, acc[1][1], false, false);
    }

    // ---- write next stage to LDS, sync ----
    if (more) {
#pragma unroll
      for (int j = 0; j < 2; ++j) {
        int q = tid + j * 256, row = q >> 3, c8 = (q & 7) * 8;
        *(bf16x8*)&SA[nxt][row * LDK + c8] = rA[j];
      }
#pragma unroll
      for (int j = 0; j < 4; ++j) {
        int q = tid + j * 256, row = q >> 3, c8 = (q & 7) * 8;
        *(bf16x8*)&SB[nxt][row * LDK + c8] = rB[j];
      }
    }
    __syncthreads();
  }

  // ---- epilogue: C/D layout: VGPR r -> row +r+8*half, col l15 ----
#pragma unroll
  for (int mi = 0; mi < 2; ++mi) {
#pragma unroll
    for (int ni = 0; ni < 2; ++ni) {
#pragma unroll
      for (int r = 0; r < 8; ++r) {
        const int mm = mb0 + wm * 32 + mi * 16 + r + 8 * half;
        const int nn = nb0 + wn * 32 + ni * 16 + l15;
        const float v = acc[mi][ni][r];
        if constexpr (MODE == 1) {
          float t  = v + bias[nn];
          float dt = (t > 20.0f) ? t : log1pf(expf(t));   // softplus
          float ae = expf(Avec[nn] * dt);
          size_t idx = (size_t)mm * N + nn;
          float xc = xconv[idx];
          s_out[idx]    = xc * dt * ae;                   // x_weighted * A_exp
          aexp_out[idx] = ae;
        } else {
          C[(size_t)mm * N + nn] = v;
        }
      }
    }
  }
}

// --------------------------------------------------- depthwise conv + silu --
__global__ void conv_silu_kernel(const float* __restrict__ xz,    // NTOK x 3072
                                 const float* __restrict__ cw,    // (1536,1,3)
                                 const float* __restrict__ cb,
                                 float* __restrict__ xconv_f32,   // NTOK x 1536
                                 __bf16* __restrict__ xconv_bf) {
  int i = blockIdx.x * blockDim.x + threadIdx.x;
  if (i >= NTOK * DINNER) return;
  int c = i % DINNER;
  int t = i / DINNER;          // global token index
  int l = t % LSEQ;
  size_t base = (size_t)t * DPROJ + c;               // x_inner cols [0,1536)
  float xm = (l > 0)        ? xz[base - DPROJ] : 0.0f;
  float x0 = xz[base];
  float xp = (l < LSEQ - 1) ? xz[base + DPROJ] : 0.0f;
  float v = cw[c * 3 + 0] * xm + cw[c * 3 + 1] * x0 + cw[c * 3 + 2] * xp + cb[c];
  float s = silu(v);
  xconv_f32[i] = s;
  xconv_bf[i]  = f2bf(s);
}

// ------------------------------------------------- chunked parallel scan ----
__global__ void scan_partial_kernel(const float* __restrict__ s_in,
                                    float* __restrict__ partial) {
  int i = blockIdx.x * blockDim.x + threadIdx.x;   // (b*NCHUNK + ch)*DINNER + c
  if (i >= NBAT * NCHUNK * DINNER) return;
  int c  = i % DINNER;
  int bc = i / DINNER;
  int ch = bc % NCHUNK;
  int b  = bc / NCHUNK;
  size_t idx = ((size_t)b * LSEQ + (size_t)ch * CHUNK) * DINNER + c;
  float sum = 0.0f;
  for (int l = 0; l < CHUNK; ++l) {
    sum += s_in[idx];
    idx += DINNER;
  }
  partial[i] = sum;
}

__global__ void scan_offsets_kernel(float* __restrict__ partial) {
  int i = blockIdx.x * blockDim.x + threadIdx.x;   // b*DINNER + c
  if (i >= NBAT * DINNER) return;
  int c = i % DINNER;
  int b = i / DINNER;
  float run = 0.0f;
  for (int ch = 0; ch < NCHUNK; ++ch) {
    size_t idx = ((size_t)b * NCHUNK + ch) * DINNER + c;
    float t = partial[idx];
    partial[idx] = run;                            // exclusive prefix
    run += t;
  }
}

__global__ void scan_apply_kernel(const float* __restrict__ s_in,
                                  const float* __restrict__ aexp,
                                  const float* __restrict__ xconv,
                                  const float* __restrict__ Dp,
                                  const float* __restrict__ xz,   // z cols
                                  const float* __restrict__ partial,
                                  __bf16* __restrict__ y_bf) {
  int i = blockIdx.x * blockDim.x + threadIdx.x;
  if (i >= NBAT * NCHUNK * DINNER) return;
  int c  = i % DINNER;
  int bc = i / DINNER;
  int ch = bc % NCHUNK;
  int b  = bc / NCHUNK;
  float run = partial[i];
  const float dp = Dp[c];
  size_t idx  = ((size_t)b * LSEQ + (size_t)ch * CHUNK) * DINNER + c;
  size_t zidx = ((size_t)b * LSEQ + (size_t)ch * CHUNK) * DPROJ + DINNER + c;
  for (int l = 0; l < CHUNK; ++l) {
    run += s_in[idx];
    float y = run * aexp[idx] + xconv[idx] * dp;
    y *= silu(xz[zidx]);
    y_bf[idx] = f2bf(y);
    idx += DINNER;
    zidx += DPROJ;
  }
}

// ---------------------------------------------------------------- launch ----
static inline char* carve(char*& p, size_t bytes) {
  char* r = p;
  p += (bytes + 255) & ~(size_t)255;
  return r;
}

extern "C" void kernel_launch(void* const* d_in, const int* in_sizes, int n_in,
                              void* d_out, int out_size, void* d_ws, size_t ws_size,
                              hipStream_t stream) {
  const float* x      = (const float*)d_in[0];   // (4,4096,1024)
  const float* W_in   = (const float*)d_in[1];   // (3072,1024)
  const float* W_out  = (const float*)d_in[2];   // (1024,1536)
  const float* conv_w = (const float*)d_in[3];   // (1536,1,3)
  const float* conv_b = (const float*)d_in[4];   // (1536,)
  const float* dt_w   = (const float*)d_in[5];   // (1536,1536)
  const float* dt_b   = (const float*)d_in[6];   // (1536,)
  const float* Avec   = (const float*)d_in[7];   // (1536,)
  const float* Dp     = (const float*)d_in[8];   // (1536,)
  float* out = (float*)d_out;                    // (4,4096,1024)

  char* p = (char*)d_ws;
  __bf16* x_bf     = (__bf16*)carve(p, (size_t)NTOK * DIMX * 2);
  __bf16* Win_bf   = (__bf16*)carve(p, (size_t)DPROJ * DIMX * 2);
  __bf16* dtw_bf   = (__bf16*)carve(p, (size_t)DINNER * DINNER * 2);
  __bf16* Wout_bf  = (__bf16*)carve(p, (size_t)DIMX * DINNER * 2);
  float*  xz       = (float*) carve(p, (size_t)NTOK * DPROJ * 4);
  float*  xconv32  = (float*) carve(p, (size_t)NTOK * DINNER * 4);
  __bf16* xconv_bf = (__bf16*)carve(p, (size_t)NTOK * DINNER * 2);
  float*  s_buf    = (float*) carve(p, (size_t)NTOK * DINNER * 4);
  float*  ae_buf   = (float*) carve(p, (size_t)NTOK * DINNER * 4);
  __bf16* y_bf     = (__bf16*)carve(p, (size_t)NTOK * DINNER * 2);
  float*  part_buf = (float*) carve(p, (size_t)NBAT * NCHUNK * DINNER * 4);

  // 1) fp32 -> bf16 conversions (activations + weights)
  {
    int n;
    n = NTOK * DIMX;
    cvt_bf16_kernel<<<(n + 255) / 256, 256, 0, stream>>>(x, x_bf, n);
    n = DPROJ * DIMX;
    cvt_bf16_kernel<<<(n + 255) / 256, 256, 0, stream>>>(W_in, Win_bf, n);
    n = DINNER * DINNER;
    cvt_bf16_kernel<<<(n + 255) / 256, 256, 0, stream>>>(dt_w, dtw_bf, n);
    n = DIMX * DINNER;
    cvt_bf16_kernel<<<(n + 255) / 256, 256, 0, stream>>>(W_out, Wout_bf, n);
  }

  // grids are exact: M=16384 -> 256 row-blocks; N/128 col-blocks
  // 2) GEMM1: xz = x @ W_in^T   (N=3072, K=1024)
  wmma_gemm_lds<0, DPROJ, DIMX><<<256 * (DPROJ / 128), 256, 0, stream>>>(
      x_bf, Win_bf, xz, nullptr, nullptr, nullptr, nullptr, nullptr);

  // 3) depthwise conv + bias + silu
  {
    int n = NTOK * DINNER;
    conv_silu_kernel<<<(n + 255) / 256, 256, 0, stream>>>(
        xz, conv_w, conv_b, xconv32, xconv_bf);
  }

  // 4) GEMM2 with fused softplus/exp epilogue (N=1536, K=1536)
  wmma_gemm_lds<1, DINNER, DINNER><<<256 * (DINNER / 128), 256, 0, stream>>>(
      xconv_bf, dtw_bf, nullptr, dt_b, Avec, xconv32, s_buf, ae_buf);

  // 5) chunked parallel scan (3 phases)
  {
    int n1 = NBAT * NCHUNK * DINNER;   // 196608
    scan_partial_kernel<<<(n1 + 255) / 256, 256, 0, stream>>>(s_buf, part_buf);
    int n2 = NBAT * DINNER;            // 6144
    scan_offsets_kernel<<<(n2 + 255) / 256, 256, 0, stream>>>(part_buf);
    scan_apply_kernel<<<(n1 + 255) / 256, 256, 0, stream>>>(
        s_buf, ae_buf, xconv32, Dp, xz, part_buf, y_bf);
  }

  // 6) GEMM3: out = y @ W_out^T  (N=1024, K=1536)
  wmma_gemm_lds<2, DIMX, DINNER><<<256 * (DIMX / 128), 256, 0, stream>>>(
      y_bf, Wout_bf, out, nullptr, nullptr, nullptr, nullptr, nullptr);

  (void)in_sizes; (void)n_in; (void)out_size; (void)ws_size;
}
